// VideoMambaVideoEncoder_60455959658705
// MI455X (gfx1250) — matrix operand order repaired and checked by reading, not compiled, not run on hardware
//
#include <hip/hip_runtime.h>
#include <math.h>

// ---------------------------------------------------------------------------
// VideoMamba video encoder forward, CDNA5 (gfx1250, wave32, WMMA).
// All dense projections run on v_wmma_f32_16x16x32_f16 (f16 in, f32 acc).
// ---------------------------------------------------------------------------

typedef __attribute__((ext_vector_type(16))) _Float16 v16h;
typedef __attribute__((ext_vector_type(8)))  float    v8f;
typedef __attribute__((ext_vector_type(4)))  float    f4;
typedef __attribute__((ext_vector_type(4)))  _Float16 h4;

#define LTOK   785      // 1 + 4*14*14 tokens
#define DM     384      // model dim
#define EDIM   768      // inner dim E
#define NSTATE 16       // SSM state N
#define RRANK  24       // dt rank R
#define DBLW   56       // R + 2N
#define NL     12       // layers
#define SCHUNK 8        // scan steps staged per barrier pair

__device__ __forceinline__ float sigmoidf_(float x) { return 1.0f / (1.0f + __expf(-x)); }

// ---------------------------------------------------------------------------
// WMMA GEMM:  C[M,N] = epi(A[M,K] @ W[N,K]^T + bias)
// Block tile 64x64, 128 threads = 4 waves.  Wave w owns N cols [16w,16w+16)
// and all four 16-row M subtiles (4 accumulators, 4 WMMAs per k-step sharing
// one B fragment).  fp32 tiles are converted to f16 in LDS.  Full interior
// tiles take a branchless vectorized load path (global_load_b128 +
// ds_store_b64 + global_prefetch); edge tiles use clamped+select loads.
// epilogue: 0=none, 1=+bias, 2=+bias then softplus.
// ---------------------------------------------------------------------------
__global__ void gemm_wmma_kernel(const float* __restrict__ A, int lda,
                                 const float* __restrict__ W,
                                 const float* __restrict__ bias,
                                 float* __restrict__ C, int ldc,
                                 int M, int N, int K, int epilogue)
{
    __shared__ __attribute__((aligned(16))) _Float16 sA[64 * 32];   // [m][k]
    __shared__ __attribute__((aligned(16))) _Float16 sB[64 * 32];   // [n][k]
    const int tid  = threadIdx.x;
    const int wave = tid >> 5;
    const int lane = tid & 31;
    const int half = lane >> 4;        // K-half selector
    const int lrow = lane & 15;        // M row (A frag) / N col (B frag)
    const int m0 = blockIdx.y << 6;
    const int n0 = blockIdx.x << 6;

    // Block-uniform fast-path predicate (depends only on blockIdx / dims).
    const bool full = (m0 + 64 <= M) && (n0 + 64 <= N) && ((K & 31) == 0);

    v8f acc[4];
#pragma unroll
    for (int mt = 0; mt < 4; ++mt)
        acc[mt] = (v8f){0.f, 0.f, 0.f, 0.f, 0.f, 0.f, 0.f, 0.f};

    for (int k0 = 0; k0 < K; k0 += 32) {
        if (full) {
            // 2048 floats per tile; each thread: 4x float4 -> half4 per tile.
#pragma unroll
            for (int r = 0; r < 4; ++r) {
                int i = (tid + (r << 7)) << 2;       // element index, mult of 4
                int m = i >> 5, k = i & 31;
                f4 av = *reinterpret_cast<const f4*>(A + (size_t)(m0 + m) * lda + (k0 + k));
                *reinterpret_cast<h4*>(&sA[i]) = __builtin_convertvector(av, h4);
                f4 wv = *reinterpret_cast<const f4*>(W + (size_t)(n0 + m) * K + (k0 + k));
                *reinterpret_cast<h4*>(&sB[i]) = __builtin_convertvector(wv, h4);
            }
            if (k0 + 32 < K) {   // prefetch next k-tile rows
                int pr = tid & 63;
                __builtin_prefetch(A + (size_t)(m0 + pr) * lda + (k0 + 32), 0, 1);
                __builtin_prefetch(W + (size_t)(n0 + pr) * K + (k0 + 32), 0, 1);
            }
        } else {
            // Edge tile: unconditional clamped loads + select (no exec branches).
            for (int i = tid; i < 64 * 32; i += 128) {
                int m = i >> 5, k = i & 31;
                int gm = m0 + m, gn = n0 + m, gk = k0 + k;
                int cm = gm < M ? gm : (M - 1);
                int cn = gn < N ? gn : (N - 1);
                int ck = gk < K ? gk : (K - 1);
                float av = A[(size_t)cm * lda + ck];
                float wv = W[(size_t)cn * K + ck];
                sA[i] = (_Float16)((gm < M && gk < K) ? av : 0.0f);
                sB[i] = (_Float16)((gn < N && gk < K) ? wv : 0.0f);
            }
        }
        __syncthreads();

        // B fragment (shared by all 4 M subtiles): 32x16, row-striped, K split
        // across lane halves (ISA 7.12.2).
        v16h bf;
#pragma unroll
        for (int j = 0; j < 8; ++j) {
            int kb = ((j >> 2) << 4) + (half << 3) + ((j & 3) << 1);
            int nn = (wave << 4) + lrow;
            bf[2 * j]     = sB[nn * 32 + kb];
            bf[2 * j + 1] = sB[nn * 32 + kb + 1];
        }
#pragma unroll
        for (int mt = 0; mt < 4; ++mt) {
            v16h af;
#pragma unroll
            for (int j = 0; j < 8; ++j) {
                int ka = ((j >> 2) << 4) + (half << 3) + ((j & 3) << 1);
                af[2 * j]     = sA[((mt << 4) + lrow) * 32 + ka];
                af[2 * j + 1] = sA[((mt << 4) + lrow) * 32 + ka + 1];
            }
            acc[mt] = __builtin_amdgcn_wmma_f32_16x16x32_f16(false, af, false, bf,
                                                             (short)0, acc[mt], false, false);
        }
        __syncthreads();
    }

    // C/D layout: col = lane&15 (+ wave tile), rows mt*16 + i + 8*half.
    int col = n0 + (wave << 4) + lrow;
    if (col < N) {
        float bv = (epilogue >= 1 && bias) ? bias[col] : 0.0f;
#pragma unroll
        for (int mt = 0; mt < 4; ++mt) {
#pragma unroll
            for (int i = 0; i < 8; ++i) {
                int row = m0 + (mt << 4) + i + (half << 3);
                if (row < M) {
                    float v = acc[mt][i] + bv;
                    if (epilogue == 2)  // softplus
                        v = (v > 20.0f) ? v : log1pf(__expf(v));
                    C[(size_t)row * ldc + col] = v;
                }
            }
        }
    }
}

// ---------------------------------------------------------------------------
// Gather video into patch rows: patches[784][768], feature = c*256 + a*16 + b.
// ---------------------------------------------------------------------------
__global__ void patchify_kernel(const float* __restrict__ x, float* __restrict__ patches)
{
    int idx = blockIdx.x * blockDim.x + threadIdx.x;
    const int NTOK = 784, F = 768;
    if (idx >= NTOK * F) return;
    int tokid = idx / F;
    int f = idx - tokid * F;
    int t = tokid / 196;
    int rem = tokid - t * 196;
    int pi = rem / 14;
    int pj = rem - pi * 14;
    int c = f >> 8;
    int ab = f & 255;
    int a = ab >> 4;
    int b = ab & 15;
    int h = pi * 16 + a;
    int w = pj * 16 + b;
    patches[idx] = x[(((size_t)c * 4 + t) * 224 + h) * 224 + w];
}

// hidden = [cls | tok] + pos ; residual = 0
__global__ void assemble_kernel(const float* __restrict__ tokg, const float* __restrict__ cls,
                                const float* __restrict__ pos, float* __restrict__ hidden,
                                float* __restrict__ residual)
{
    int idx = blockIdx.x * blockDim.x + threadIdx.x;
    if (idx >= LTOK * DM) return;
    int r = idx / DM;
    int d = idx - r * DM;
    float v = (r == 0) ? cls[d] : tokg[(size_t)(r - 1) * DM + d];
    hidden[idx]   = v + pos[idx];
    residual[idx] = 0.0f;
}

// residual += hidden; out = rmsnorm(residual) * w.  One block per token row.
__global__ void add_rmsnorm_kernel(const float* __restrict__ hidden, float* __restrict__ residual,
                                   const float* __restrict__ w, float* __restrict__ out)
{
    __shared__ float red[128];
    int row = blockIdx.x;
    int tid = threadIdx.x;
    size_t base = (size_t)row * DM;
    float vals[3];
    float ss = 0.f;
#pragma unroll
    for (int i = 0; i < 3; ++i) {
        int d = tid + i * 128;
        float r = residual[base + d] + hidden[base + d];
        residual[base + d] = r;
        vals[i] = r;
        ss += r * r;
    }
    red[tid] = ss;
    __syncthreads();
    for (int s = 64; s > 0; s >>= 1) {
        if (tid < s) red[tid] += red[tid + s];
        __syncthreads();
    }
    float scale = rsqrtf(red[0] / (float)DM + 1e-5f);
#pragma unroll
    for (int i = 0; i < 3; ++i) {
        int d = tid + i * 128;
        out[base + d] = vals[i] * scale * w[d];
    }
}

// Depthwise causal conv (K=4) + silu.  dir=0: taps t-3..t; dir=1 (reversed
// sequence, un-reversed indexing): taps t..t+3 with reversed kernel order.
__global__ void dwconv_silu_kernel(const float* __restrict__ xz, const float* __restrict__ cw,
                                   const float* __restrict__ cb, float* __restrict__ xs, int dir)
{
    int idx = blockIdx.x * blockDim.x + threadIdx.x;
    if (idx >= LTOK * EDIM) return;
    int t = idx / EDIM;
    int e = idx - t * EDIM;
    float acc = cb[e];
#pragma unroll
    for (int k = 0; k < 4; ++k) {
        int tt = dir ? (t + 3 - k) : (t - 3 + k);
        if (tt >= 0 && tt < LTOK)
            acc += cw[e * 4 + k] * xz[(size_t)tt * (2 * EDIM) + e];
    }
    xs[idx] = acc * sigmoidf_(acc);
}

// Selective-scan: lane owns channel e, h[16] state in registers.  B/C coeffs
// (uniform across channels) are staged SCHUNK time-steps at a time through
// LDS so we pay one barrier pair per 8 steps instead of per step.
__global__ void ssm_scan_kernel(const float* __restrict__ xs, const float* __restrict__ dt,
                                const float* __restrict__ dbl, const float* __restrict__ Alog,
                                const float* __restrict__ Dp, float* __restrict__ y, int dir)
{
    __shared__ float sBC[SCHUNK][32];
    int e = blockIdx.x * blockDim.x + threadIdx.x;   // 3 blocks * 256 = 768
    int tid = threadIdx.x;
    float Ae[NSTATE], h[NSTATE];
#pragma unroll
    for (int n = 0; n < NSTATE; ++n) {
        Ae[n] = -__expf(Alog[(size_t)e * NSTATE + n]);
        h[n] = 0.0f;
    }
    float De = Dp[e];
    const int cstage = tid >> 5;    // 0..7 (256 threads = 8 x 32)
    const int ccol   = tid & 31;
    for (int s0 = 0; s0 < LTOK; s0 += SCHUNK) {
        {   // stage B|C for up to SCHUNK steps
            int s = s0 + cstage;
            if (s < LTOK) {
                int t = dir ? (LTOK - 1 - s) : s;
                sBC[cstage][ccol] = dbl[(size_t)t * DBLW + RRANK + ccol];
            }
        }
        __syncthreads();
        int cend = (LTOK - s0) < SCHUNK ? (LTOK - s0) : SCHUNK;
        for (int cc = 0; cc < cend; ++cc) {
            int s = s0 + cc;
            int t = dir ? (LTOK - 1 - s) : s;
            float dte = dt[(size_t)t * EDIM + e];
            float xte = xs[(size_t)t * EDIM + e];
            float dx = dte * xte;
            float acc = 0.0f;
#pragma unroll
            for (int n = 0; n < NSTATE; ++n) {
                h[n] = __expf(dte * Ae[n]) * h[n] + dx * sBC[cc][n];
                acc += h[n] * sBC[cc][16 + n];
            }
            y[(size_t)t * EDIM + e] = acc + xte * De;
        }
        __syncthreads();
    }
}

// comb = (yf + yb) * silu(z)
__global__ void gate_combine_kernel(const float* __restrict__ yf, const float* __restrict__ yb,
                                    const float* __restrict__ xz, float* __restrict__ comb)
{
    int idx = blockIdx.x * blockDim.x + threadIdx.x;
    if (idx >= LTOK * EDIM) return;
    int t = idx / EDIM;
    int e = idx - t * EDIM;
    float z = xz[(size_t)t * (2 * EDIM) + EDIM + e];
    comb[idx] = (yf[idx] + yb[idx]) * z * sigmoidf_(z);
}

// ---------------------------------------------------------------------------
extern "C" void kernel_launch(void* const* d_in, const int* in_sizes, int n_in,
                              void* d_out, int out_size, void* d_ws, size_t ws_size,
                              hipStream_t stream)
{
    (void)in_sizes; (void)n_in; (void)out_size; (void)ws_size;

    const float* x         = (const float*)d_in[0];
    const float* patch_w   = (const float*)d_in[1];
    const float* patch_b   = (const float*)d_in[2];
    const float* cls       = (const float*)d_in[3];
    const float* pos       = (const float*)d_in[4];
    const float* norm_w    = (const float*)d_in[5];
    const float* in_w      = (const float*)d_in[6];
    const float* conv_w    = (const float*)d_in[7];
    const float* conv_b    = (const float*)d_in[8];
    const float* xproj_w   = (const float*)d_in[9];
    const float* dtproj_w  = (const float*)d_in[10];
    const float* dtproj_b  = (const float*)d_in[11];
    const float* A_log     = (const float*)d_in[12];
    const float* Dp        = (const float*)d_in[13];
    const float* conv_w_b  = (const float*)d_in[14];
    const float* conv_b_b  = (const float*)d_in[15];
    const float* xproj_wb  = (const float*)d_in[16];
    const float* dtproj_wb = (const float*)d_in[17];
    const float* dtproj_bb = (const float*)d_in[18];
    const float* A_b_log   = (const float*)d_in[19];
    const float* D_b       = (const float*)d_in[20];
    const float* out_w     = (const float*)d_in[21];
    const float* norm_f_w  = (const float*)d_in[22];

    float* ws = (float*)d_ws;
    float* patches  = ws; ws += 784 * 768;
    float* tokg     = ws; ws += 784 * 384;
    float* hidden   = ws; ws += LTOK * DM;
    float* residual = ws; ws += LTOK * DM;
    float* normed   = ws; ws += LTOK * DM;
    float* xz       = ws; ws += LTOK * 2 * EDIM;
    float* xs_f     = ws; ws += LTOK * EDIM;
    float* xs_b     = ws; ws += LTOK * EDIM;
    float* dbl_f    = ws; ws += LTOK * DBLW;
    float* dbl_b    = ws; ws += LTOK * DBLW;
    float* dt_f     = ws; ws += LTOK * EDIM;
    float* dt_b     = ws; ws += LTOK * EDIM;
    float* y_f      = ws; ws += LTOK * EDIM;
    float* y_b      = ws; ws += LTOK * EDIM;
    float* comb     = ws; ws += LTOK * EDIM;

    auto gemm = [&](const float* A, int lda, const float* W, const float* bias,
                    float* C, int ldc, int M, int N, int K, int epi) {
        dim3 grid((N + 63) / 64, (M + 63) / 64);
        gemm_wmma_kernel<<<grid, 128, 0, stream>>>(A, lda, W, bias, C, ldc, M, N, K, epi);
    };

    {   // Patch embedding
        int n = 784 * 768;
        patchify_kernel<<<(n + 255) / 256, 256, 0, stream>>>(x, patches);
    }
    gemm(patches, 768, patch_w, patch_b, tokg, 384, 784, 384, 768, 1);
    {
        int n = LTOK * DM;
        assemble_kernel<<<(n + 255) / 256, 256, 0, stream>>>(tokg, cls, pos, hidden, residual);
    }

    const int nLE = LTOK * EDIM;
    for (int l = 0; l < NL; ++l) {
        const float* nw  = norm_w    + (size_t)l * DM;
        const float* iw  = in_w      + (size_t)l * 2 * EDIM * DM;
        const float* cwf = conv_w    + (size_t)l * EDIM * 4;
        const float* cbf = conv_b    + (size_t)l * EDIM;
        const float* xpf = xproj_w   + (size_t)l * DBLW * EDIM;
        const float* dwf = dtproj_w  + (size_t)l * EDIM * RRANK;
        const float* dbf = dtproj_b  + (size_t)l * EDIM;
        const float* Alf = A_log     + (size_t)l * EDIM * NSTATE;
        const float* Dpf = Dp        + (size_t)l * EDIM;
        const float* cwb = conv_w_b  + (size_t)l * EDIM * 4;
        const float* cbb = conv_b_b  + (size_t)l * EDIM;
        const float* xpb = xproj_wb  + (size_t)l * DBLW * EDIM;
        const float* dwb = dtproj_wb + (size_t)l * EDIM * RRANK;
        const float* dbb = dtproj_bb + (size_t)l * EDIM;
        const float* Alb = A_b_log   + (size_t)l * EDIM * NSTATE;
        const float* Dpb = D_b       + (size_t)l * EDIM;
        const float* ow  = out_w     + (size_t)l * DM * EDIM;

        // residual += hidden; normed = rmsnorm(residual)
        add_rmsnorm_kernel<<<LTOK, 128, 0, stream>>>(hidden, residual, nw, normed);

        // xz = normed @ in_w^T   [785 x 1536]
        gemm(normed, DM, iw, nullptr, xz, 2 * EDIM, LTOK, 2 * EDIM, DM, 0);

        // depthwise conv + silu, both directions
        dwconv_silu_kernel<<<(nLE + 255) / 256, 256, 0, stream>>>(xz, cwf, cbf, xs_f, 0);
        dwconv_silu_kernel<<<(nLE + 255) / 256, 256, 0, stream>>>(xz, cwb, cbb, xs_b, 1);

        // dbl = xs @ xproj^T   [785 x 56]
        gemm(xs_f, EDIM, xpf, nullptr, dbl_f, DBLW, LTOK, DBLW, EDIM, 0);
        gemm(xs_b, EDIM, xpb, nullptr, dbl_b, DBLW, LTOK, DBLW, EDIM, 0);

        // dt = softplus(dtr @ dtw^T + dtb)   (dtr = dbl[:, :24], lda = 56)
        gemm(dbl_f, DBLW, dwf, dbf, dt_f, EDIM, LTOK, EDIM, RRANK, 2);
        gemm(dbl_b, DBLW, dwb, dbb, dt_b, EDIM, LTOK, EDIM, RRANK, 2);

        // selective scans (forward / reversed time)
        ssm_scan_kernel<<<3, 256, 0, stream>>>(xs_f, dt_f, dbl_f, Alf, Dpf, y_f, 0);
        ssm_scan_kernel<<<3, 256, 0, stream>>>(xs_b, dt_b, dbl_b, Alb, Dpb, y_b, 1);

        // (yf + yb) * silu(z)
        gate_combine_kernel<<<(nLE + 255) / 256, 256, 0, stream>>>(y_f, y_b, xz, comb);

        // hidden = comb @ out_w^T   [785 x 384]
        gemm(comb, EDIM, ow, nullptr, hidden, DM, LTOK, DM, EDIM, 0);
    }

    // out = rmsnorm(residual + hidden) * norm_f_w
    add_rmsnorm_kernel<<<LTOK, 128, 0, stream>>>(hidden, residual, norm_f_w, (float*)d_out);
}
